// MultiHeadAttention_61211873903529
// MI455X (gfx1250) — matrix operand order repaired and checked by reading, not compiled, not run on hardware
//
#include <hip/hip_runtime.h>
#include <hip/hip_bf16.h>

#define D_MODEL 1024
#define N_HEADS 16
#define DH      64
#define BATCH   2
#define SEQ     2048
#define M_TOT   (BATCH*SEQ)          // 4096
#define N_QKV   (3*D_MODEL)          // 3072

typedef __attribute__((ext_vector_type(16))) __bf16 v16bf;
typedef __attribute__((ext_vector_type(8)))  __bf16 v8bf;
typedef __attribute__((ext_vector_type(4)))  __bf16 v4bf;
typedef __attribute__((ext_vector_type(8)))  float  v8f;
typedef __attribute__((ext_vector_type(4)))  float  v4f;

// ---------------------------------------------------------------------------
// WMMA fragment loaders (wave32 layouts per CDNA5 ISA §7.12.2)
// A-matrix 16x32 bf16: lane m = L&15; lo lanes hold K 0..7 & 16..23,
//                      hi lanes hold K 8..15 & 24..31 (2 K per VGPR).
// B-matrix 32x16 bf16: lane n = L&15; lo lanes hold K 0..7 & 16..23,
//                      hi lanes hold K 8..15 & 24..31 (2 K per VGPR).
// Works for both global and LDS source pointers.
// ---------------------------------------------------------------------------
static __device__ __forceinline__ v16bf load_a_frag(const __bf16* A, int lda,
                                                    int m0, int kk) {
  const int lane = threadIdx.x & 31;
  const int mp   = lane & 15;
  const int hi   = lane >> 4;
  const __bf16* p = A + (size_t)(m0 + mp) * lda + kk + hi * 8;
  v8bf a = *(const v8bf*)p;          // K = kk + (hi?8:0) + 0..7
  v8bf b = *(const v8bf*)(p + 16);   // K = kk + 16 + (hi?8:0) + 0..7
  v16bf r;
#pragma unroll
  for (int i = 0; i < 8; ++i) { r[i] = a[i]; r[i + 8] = b[i]; }
  return r;
}

static __device__ __forceinline__ v16bf load_b_frag(const __bf16* B, int ldb,
                                                    int n0, int kk) {
  const int lane = threadIdx.x & 31;
  const int np   = lane & 15;
  const int hi   = lane >> 4;
  const __bf16* p = B + (size_t)(n0 + np) * ldb + kk + hi * 8;
  v8bf a = *(const v8bf*)p;          // K = kk + (hi?8:0) + 0..7
  v8bf b = *(const v8bf*)(p + 16);   // K = kk + 16 + (hi?8:0) + 0..7
  v16bf r;
#pragma unroll
  for (int i = 0; i < 8; ++i) { r[i] = a[i]; r[i + 8] = b[i]; }
  return r;
}

// ---------------------------------------------------------------------------
// fp32 -> bf16 conversion (4-wide)
// ---------------------------------------------------------------------------
__global__ void cvt_f32_bf16(const float* __restrict__ in,
                             __bf16* __restrict__ out, int n4) {
  int i = blockIdx.x * blockDim.x + threadIdx.x;
  if (i < n4) {
    v4f v = ((const v4f*)in)[i];
    v4bf o;
    o[0] = (__bf16)v[0]; o[1] = (__bf16)v[1];
    o[2] = (__bf16)v[2]; o[3] = (__bf16)v[3];
    ((v4bf*)out)[i] = o;
  }
}

// ---------------------------------------------------------------------------
// C[M,N] = A[M,K] @ W[N,K]^T  (nn.Linear), bf16 in, f32 accumulate.
// Block-tiled: 256 threads = 8 waves (4 M x 2 N), 128x128 tile, K-step 32.
// LDS double-buffered (register prefetch, one barrier per K-step).
// Per K-step: 16 KB global traffic for 1 MFLOP -> 64 FLOP/byte from HBM/L2;
// all WMMA fragment reuse is served by LDS (rows padded to 80 B stride).
// ---------------------------------------------------------------------------
template <typename OutT>
__global__ __launch_bounds__(256) void gemm_wmma(const __bf16* __restrict__ A,
                                                 const __bf16* __restrict__ W,
                                                 OutT* __restrict__ C,
                                                 int M, int N, int K) {
  constexpr int LDT = 40;                       // padded LDS row stride (elems)
  __shared__ __align__(16) __bf16 lds[2][2][128 * LDT];   // 40 KB
  const int tid  = threadIdx.x;
  const int wave = tid >> 5;
  const int lane = tid & 31;
  const int np   = lane & 15;
  const int hi   = lane >> 4;
  const int wm   = wave >> 1;                   // 0..3 (M)
  const int wn   = wave & 1;                    // 0..1 (N)
  const int m0 = blockIdx.x * 128;
  const int n0 = blockIdx.y * 128;

  // cooperative tile load: 128 rows x 32 cols bf16 = 512 chunks of 16B,
  // 2 chunks per thread per tile
  const int c0 = tid, c1 = tid + 256;
  const int r0 = c0 >> 2, s0 = (c0 & 3) * 8;
  const int r1 = c1 >> 2, s1 = (c1 & 3) * 8;

  auto gload = [&](const __bf16* src, int base, int kks, v8bf& x, v8bf& y) {
    x = *(const v8bf*)(src + (size_t)(base + r0) * K + kks + s0);
    y = *(const v8bf*)(src + (size_t)(base + r1) * K + kks + s1);
  };
  auto stTile = [&](int buf, v8bf ax, v8bf ay, v8bf bx, v8bf by) {
    *(v8bf*)(&lds[buf][0][r0 * LDT + s0]) = ax;
    *(v8bf*)(&lds[buf][0][r1 * LDT + s1]) = ay;
    *(v8bf*)(&lds[buf][1][r0 * LDT + s0]) = bx;
    *(v8bf*)(&lds[buf][1][r1 * LDT + s1]) = by;
  };

  v8f zero = {};
  v8f acc[2][4];
#pragma unroll
  for (int i = 0; i < 2; ++i)
#pragma unroll
    for (int t = 0; t < 4; ++t) acc[i][t] = zero;

  v8bf ax, ay, bx, by;
  gload(A, m0, 0, ax, ay);
  gload(W, n0, 0, bx, by);
  stTile(0, ax, ay, bx, by);
  __syncthreads();

  for (int kks = 0; kks < K; kks += 32) {
    const int  cur  = (kks >> 5) & 1;
    const bool more = (kks + 32) < K;
    if (more) {                                  // prefetch next K-slice
      gload(A, m0, kks + 32, ax, ay);
      gload(W, n0, kks + 32, bx, by);
    }
    const __bf16* la = lds[cur][0];
    const __bf16* lb = lds[cur][1];
    v16bf a0 = load_a_frag(la, LDT, wm * 32,      0);
    v16bf a1 = load_a_frag(la, LDT, wm * 32 + 16, 0);
#pragma unroll
    for (int t = 0; t < 4; ++t) {
      v16bf b = load_b_frag(lb, LDT, wn * 64 + 16 * t, 0);
      acc[0][t] = __builtin_amdgcn_wmma_f32_16x16x32_bf16(
          false, a0, false, b, (short)0, acc[0][t], false, false);
      acc[1][t] = __builtin_amdgcn_wmma_f32_16x16x32_bf16(
          false, a1, false, b, (short)0, acc[1][t], false, false);
    }
    if (more) {                                  // fill the other buffer
      stTile(cur ^ 1, ax, ay, bx, by);
      __syncthreads();
    }
  }

#pragma unroll
  for (int i = 0; i < 2; ++i)
#pragma unroll
    for (int t = 0; t < 4; ++t)
#pragma unroll
      for (int r = 0; r < 8; ++r) {
        int m = m0 + wm * 32 + i * 16 + r + 8 * hi;  // C: row = vgpr + 8*half
        int n = n0 + wn * 64 + 16 * t + np;          //    col = lane & 15
        C[(size_t)m * N + n] = (OutT)acc[i][t][r];
      }
}

// ---------------------------------------------------------------------------
// Split qkv[M, 3D] into head-major Q,K [B,H,T,dh] and transposed V [B,H,dh,T]
// ---------------------------------------------------------------------------
__global__ void split_qkv(const __bf16* __restrict__ qkv,
                          __bf16* __restrict__ q, __bf16* __restrict__ k,
                          __bf16* __restrict__ vt) {
  int i = blockIdx.x * blockDim.x + threadIdx.x;   // over B*T*D
  int d = i & 63;
  int h = (i >> 6) & 15;
  int t = (i >> 10) & 2047;
  int b = i >> 21;
  size_t m  = (size_t)b * SEQ + t;
  size_t hm = (((size_t)b * N_HEADS + h) * SEQ + t) * DH + d;
  const __bf16* row = qkv + m * N_QKV + h * DH + d;
  q[hm] = row[0];
  k[hm] = row[D_MODEL];
  vt[(((size_t)b * N_HEADS + h) * DH + d) * SEQ + t] = row[2 * D_MODEL];
}

// ---------------------------------------------------------------------------
// Causal flash attention. grid.x = B*H*(SEQ/64); 4 waves/block, each wave
// owns 16 query rows and streams 32 keys per iteration with online softmax.
// ---------------------------------------------------------------------------
__global__ __launch_bounds__(128) void flash_attn(const __bf16* __restrict__ Q,
                                                  const __bf16* __restrict__ Kh,
                                                  const __bf16* __restrict__ Vt,
                                                  __bf16* __restrict__ Out) {
  const int tilesPerHead = SEQ / 64;                // 32
  const int bh   = blockIdx.x / tilesPerHead;
  const int tile = blockIdx.x % tilesPerHead;
  const int b = bh / N_HEADS, h = bh % N_HEADS;
  const int wave = threadIdx.x >> 5;
  const int lane = threadIdx.x & 31;
  const int np   = lane & 15;
  const int hi   = lane >> 4;
  const int q0 = tile * 64 + wave * 16;

  const __bf16* Qh = Q  + (size_t)bh * SEQ * DH;
  const __bf16* KH = Kh + (size_t)bh * SEQ * DH;
  const __bf16* VT = Vt + (size_t)bh * DH * SEQ;

  // Q fragments for the two 32-wide K-dim steps of dh=64
  v16bf qf0 = load_a_frag(Qh, DH, q0, 0);
  v16bf qf1 = load_a_frag(Qh, DH, q0, 32);

  v8f zero = {};
  v8f o[4];
#pragma unroll
  for (int t = 0; t < 4; ++t) o[t] = zero;
  float mrun[8], lrun[8];
#pragma unroll
  for (int r = 0; r < 8; ++r) { mrun[r] = -__builtin_inff(); lrun[r] = 0.f; }

  __shared__ __align__(16) __bf16 plds[4][16 * 32];  // per-wave P staging
  __bf16* P = plds[wave];
  const float scale = 0.125f;                        // 1/sqrt(64)

  const int kend = q0 + 16;                          // causal horizon
  for (int kk = 0; kk < kend; kk += 32) {
    // ---- S = Q K^T : two 16x16 key tiles, 2 WMMA each over dh ----
    v8f s0 = zero, s1 = zero;
    {
      v16bf b00 = load_b_frag(KH, DH, kk,      0);
      v16bf b01 = load_b_frag(KH, DH, kk,      32);
      v16bf b10 = load_b_frag(KH, DH, kk + 16, 0);
      v16bf b11 = load_b_frag(KH, DH, kk + 16, 32);
      s0 = __builtin_amdgcn_wmma_f32_16x16x32_bf16(false, qf0, false, b00,
                                                   (short)0, s0, false, false);
      s0 = __builtin_amdgcn_wmma_f32_16x16x32_bf16(false, qf1, false, b01,
                                                   (short)0, s0, false, false);
      s1 = __builtin_amdgcn_wmma_f32_16x16x32_bf16(false, qf0, false, b10,
                                                   (short)0, s1, false, false);
      s1 = __builtin_amdgcn_wmma_f32_16x16x32_bf16(false, qf1, false, b11,
                                                   (short)0, s1, false, false);
    }

    // ---- mask + online softmax; rows live across a 16-lane half ----
#pragma unroll
    for (int r = 0; r < 8; ++r) {
      const int qg = q0 + r + 8 * hi;
      float v0 = s0[r] * scale;
      float v1 = s1[r] * scale;
      if (kk + np      > qg) v0 = -__builtin_inff();
      if (kk + 16 + np > qg) v1 = -__builtin_inff();
      float mx = fmaxf(v0, v1);
      mx = fmaxf(mx, __shfl_xor(mx, 1, 16));
      mx = fmaxf(mx, __shfl_xor(mx, 2, 16));
      mx = fmaxf(mx, __shfl_xor(mx, 4, 16));
      mx = fmaxf(mx, __shfl_xor(mx, 8, 16));
      float mn = fmaxf(mrun[r], mx);
      float e0 = __expf(v0 - mn);
      float e1 = __expf(v1 - mn);
      float ps = e0 + e1;
      ps += __shfl_xor(ps, 1, 16);
      ps += __shfl_xor(ps, 2, 16);
      ps += __shfl_xor(ps, 4, 16);
      ps += __shfl_xor(ps, 8, 16);
      float c = __expf(mrun[r] - mn);
      lrun[r] = lrun[r] * c + ps;
      mrun[r] = mn;
      P[(r + 8 * hi) * 32 + np]      = (__bf16)e0;   // C-layout -> row-major LDS
      P[(r + 8 * hi) * 32 + 16 + np] = (__bf16)e1;
#pragma unroll
      for (int t = 0; t < 4; ++t) o[t][r] *= c;
    }
    asm volatile("" ::: "memory");                    // order LDS store/load
    v16bf pf = load_a_frag((const __bf16*)P, 32, 0, 0);

    // ---- O += P V : K-dim = 32 keys, N = dh (4 tiles of 16) ----
#pragma unroll
    for (int t = 0; t < 4; ++t) {
      v16bf vf = load_b_frag(VT, SEQ, 16 * t, kk);    // Vt row contiguous in keys
      o[t] = __builtin_amdgcn_wmma_f32_16x16x32_bf16(false, pf, false, vf,
                                                     (short)0, o[t], false, false);
    }
  }

  // ---- normalize and scatter back to [B,T,H*dh] bf16 ----
#pragma unroll
  for (int r = 0; r < 8; ++r) {
    float inv = 1.0f / lrun[r];
    const int qg = q0 + r + 8 * hi;
    size_t row = ((size_t)b * SEQ + qg) * D_MODEL + h * DH;
#pragma unroll
    for (int t = 0; t < 4; ++t)
      Out[row + 16 * t + np] = (__bf16)(o[t][r] * inv);
  }
}

// ---------------------------------------------------------------------------
extern "C" void kernel_launch(void* const* d_in, const int* in_sizes, int n_in,
                              void* d_out, int out_size, void* d_ws,
                              size_t ws_size, hipStream_t stream) {
  const float* x    = (const float*)d_in[0];   // [2,2048,1024]
  const float* Win  = (const float*)d_in[1];   // [3072,1024]
  const float* Wout = (const float*)d_in[2];   // [1024,1024]
  float* out = (float*)d_out;                  // [2,2048,1024]

  char* ws = (char*)d_ws;                      // layout (64 MB total):
  __bf16* xb    = (__bf16*)(ws);               //  0 MB: x bf16        (8 MB)
  __bf16* winb  = (__bf16*)(ws + (8u  << 20)); //  8 MB: W_in bf16     (6 MB)
  __bf16* woutb = (__bf16*)(ws + (14u << 20)); // 14 MB: W_out bf16    (2 MB)
  __bf16* qkvb  = (__bf16*)(ws + (16u << 20)); // 16 MB: qkv bf16     (24 MB)
  __bf16* q_hm  = (__bf16*)(ws + (40u << 20)); // 40 MB: Q head-major  (8 MB)
  __bf16* k_hm  = (__bf16*)(ws + (48u << 20)); // 48 MB: K head-major  (8 MB)
  __bf16* vt    = (__bf16*)(ws + (56u << 20)); // 56 MB: V transposed  (8 MB)
  __bf16* attnb = xb;                          // reuse: x dead after GEMM1

  // 1) fp32 -> bf16
  cvt_f32_bf16<<<(M_TOT * D_MODEL / 4 + 255) / 256, 256, 0, stream>>>(
      x, xb, M_TOT * D_MODEL / 4);
  cvt_f32_bf16<<<(N_QKV * D_MODEL / 4 + 255) / 256, 256, 0, stream>>>(
      Win, winb, N_QKV * D_MODEL / 4);
  cvt_f32_bf16<<<(D_MODEL * D_MODEL / 4 + 255) / 256, 256, 0, stream>>>(
      Wout, woutb, D_MODEL * D_MODEL / 4);

  // 2) qkv = x @ W_in^T   (bf16 out)
  gemm_wmma<__bf16><<<dim3(M_TOT / 128, N_QKV / 128), 256, 0, stream>>>(
      xb, winb, qkvb, M_TOT, N_QKV, D_MODEL);

  // 3) split into head-major Q, K and transposed V
  split_qkv<<<(M_TOT * D_MODEL) / 256, 256, 0, stream>>>(qkvb, q_hm, k_hm, vt);

  // 4) causal flash attention -> attnb [B,T,D] bf16
  flash_attn<<<BATCH * N_HEADS * (SEQ / 64), 128, 0, stream>>>(q_hm, k_hm, vt,
                                                               attnb);

  // 5) out = attn @ W_out^T  (fp32 out)
  gemm_wmma<float><<<dim3(M_TOT / 128, D_MODEL / 128), 256, 0, stream>>>(
      attnb, woutb, out, M_TOT, D_MODEL, D_MODEL);
}
